// Encoder_77996606095381
// MI455X (gfx1250) — matrix operand order repaired and checked by reading, not compiled, not run on hardware
//
#include <hip/hip_runtime.h>

// ---------------- constants (match reference) ----------------
constexpr int   HEADS     = 4;
constexpr int   HIDC      = 128;
constexpr float NEGSLOPE  = 0.2f;
constexpr float BNEPS     = 1e-5f;

typedef __attribute__((ext_vector_type(16))) __bf16        v16bf;
typedef __attribute__((ext_vector_type(8)))  float         v8f;
typedef __attribute__((ext_vector_type(8)))  unsigned int  u32x8;

// ---------------- helpers ----------------
__device__ __forceinline__ unsigned short f2bf(float f) {
    unsigned u = __float_as_uint(f);
    u += 0x7FFFu + ((u >> 16) & 1u);      // round-to-nearest-even
    return (unsigned short)(u >> 16);
}

__device__ __forceinline__ void atomicMaxF(float* addr, float val) {
    // sign-aware int trick (works with -1e30 init)
    if (val >= 0.f) atomicMax((int*)addr, __float_as_int(val));
    else            atomicMin((unsigned int*)addr, __float_as_uint(val));
}

// ---------------- utility kernels ----------------
__global__ void k_fill(float* p, float v, long n) {
    long i = (long)blockIdx.x * blockDim.x + threadIdx.x;
    if (i < n) p[i] = v;
}

__global__ void k_f32_to_bf16(const float* __restrict__ in,
                              unsigned short* __restrict__ out, long n) {
    long i = (long)blockIdx.x * blockDim.x + threadIdx.x;
    if (i < n) out[i] = f2bf(in[i]);
}

// W is K x Ncol (row-major, as in reference); Wt is Ncol x K (B column-major)
__global__ void k_transpose_bf16(const float* __restrict__ W,
                                 unsigned short* __restrict__ Wt, int K, int Ncol) {
    long i = (long)blockIdx.x * blockDim.x + threadIdx.x;
    if (i >= (long)K * Ncol) return;
    int k = (int)(i / Ncol), n = (int)(i % Ncol);
    Wt[(long)n * K + k] = f2bf(W[i]);
}

// ---------------- WMMA GEMM: C[MxNcol] = A_bf16[MxK] @ B + bias ----------------
// Bt = Ncol x K (column-major B). One wave per 16x64 C strip (4 tiles along N).
// K is a template parameter (256 or 128): all fragment offsets (j*16*K + k-step)
// become compile-time constants that fit the signed 24-bit instruction offset,
// so A and B each need only ONE address VGPR pair -> no spills, clause-grouped
// b128 loads with immediate offsets, software-pipelined with partial waits.
constexpr int NBLK = 4;   // N tiles per wave (Ncol % 64 == 0)

template<int K>
__global__ void k_wmma_gemm(const unsigned short* __restrict__ A,
                            const unsigned short* __restrict__ Bt,
                            const float* __restrict__ bias,
                            float* __restrict__ C, int M, int Ncol) {
    int wave = blockIdx.x * (blockDim.x >> 5) + (threadIdx.x >> 5);
    int lane = threadIdx.x & 31;
    int strips = Ncol >> 6;                 // Ncol / (16*NBLK)
    int tiles  = (M >> 4) * strips;
    if (wave >= tiles) return;
    int m0 = (wave / strips) << 4;
    int n0 = (wave % strips) << 6;
    int lo = lane & 15, hi = lane >> 4;

    // single address register per matrix; everything else is immediate offsets
    const unsigned short* ap = A  + (long)(m0 + lo) * K;            // A row
    const unsigned short* bp = Bt + (long)(n0 + lo) * K + 16 * hi;  // B col j=0

    // A fragment, contiguous k layout: pairs (2*p), +16*hi (matches B)
    auto loadA = [&](int k0) -> v16bf {
        u32x8 av;
#pragma unroll
        for (int p = 0; p < 8; ++p) {
            int kp = 2 * p + 16 * hi;
            av[p] = *(const unsigned int*)(ap + k0 + kp);
        }
        return __builtin_bit_cast(v16bf, av);
    };
    auto loadB = [&](int j, int k0) -> v16bf {
        return *(const v16bf*)(bp + j * 16 * K + k0);   // j*16*K+k0: compile-time
    };

    v8f c[NBLK];
#pragma unroll
    for (int j = 0; j < NBLK; ++j) c[j] = (v8f){};

    v16bf a0, a1, b0[NBLK], b1[NBLK];
    // prologue: stage 0 fragments
    a0 = loadA(0);
#pragma unroll
    for (int j = 0; j < NBLK; ++j) b0[j] = loadB(j, 0);

#pragma unroll
    for (int k0 = 0; k0 < K; k0 += 64) {
        // prefetch stage k0+32 while computing stage k0
        a1 = loadA(k0 + 32);
#pragma unroll
        for (int j = 0; j < NBLK; ++j) b1[j] = loadB(j, k0 + 32);
#pragma unroll
        for (int j = 0; j < NBLK; ++j)
            c[j] = __builtin_amdgcn_wmma_f32_16x16x32_bf16(false, a0, false, b0[j],
                                                           (short)0, c[j], false, false);
        // prefetch next iteration's stage k0+64 while computing stage k0+32
        if (k0 + 64 < K) {
            a0 = loadA(k0 + 64);
#pragma unroll
            for (int j = 0; j < NBLK; ++j) b0[j] = loadB(j, k0 + 64);
        }
#pragma unroll
        for (int j = 0; j < NBLK; ++j)
            c[j] = __builtin_amdgcn_wmma_f32_16x16x32_bf16(false, a1, false, b1[j],
                                                           (short)0, c[j], false, false);
    }

    // C layout: VGPR i -> row m0+i (+8 for upper lane half), col = n0 + 16j + lo
#pragma unroll
    for (int j = 0; j < NBLK; ++j) {
        int col = n0 + j * 16 + lo;
        float bi = bias[col];
#pragma unroll
        for (int i = 0; i < 8; ++i) {
            int row = m0 + i + 8 * hi;
            C[(long)row * Ncol + col] = c[j][i] + bi;
        }
    }
}

// ---------------- edge phase ----------------
// One wave per (edge, head): e = sum_c leaky(xl[s]+xr[d]) * att ; segment max via atomics
__global__ void k_edge_logits(const float* __restrict__ xl, const float* __restrict__ xr,
                              const int* __restrict__ src, const int* __restrict__ dst,
                              const float* __restrict__ att,
                              float* __restrict__ ebuf, float* __restrict__ ndmax,
                              int E, int N) {
    int gw   = blockIdx.x * (blockDim.x >> 5) + (threadIdx.x >> 5);
    int lane = threadIdx.x & 31;
    long tot = (long)(E + N) * HEADS;
    if (gw >= tot) return;
    int e = gw >> 2, h = gw & 3;
    int s, d;
    if (e < E) { s = src[e]; d = dst[e]; } else { s = e - E; d = s; }
    const float* pl = xl + ((long)s * HEADS + h) * HIDC;
    const float* pr = xr + ((long)d * HEADS + h) * HIDC;
    const float* pa = att + h * HIDC;
    float acc = 0.f;
#pragma unroll
    for (int c = lane; c < HIDC; c += 32) {
        float m = pl[c] + pr[c];
        m = m > 0.f ? m : m * NEGSLOPE;
        acc += m * pa[c];
    }
#pragma unroll
    for (int off = 16; off > 0; off >>= 1) acc += __shfl_xor(acc, off, 32);
    if (lane == 0) {
        ebuf[gw] = acc;
        atomicMaxF(ndmax + (long)d * HEADS + h, acc);
    }
}

__global__ void k_edge_exp(float* __restrict__ ebuf, const float* __restrict__ ndmax,
                           float* __restrict__ denom, const int* __restrict__ dst,
                           int E, int N) {
    long i = (long)blockIdx.x * blockDim.x + threadIdx.x;
    long tot = (long)(E + N) * HEADS;
    if (i >= tot) return;
    int e = (int)(i >> 2), h = (int)(i & 3);
    int d = (e < E) ? dst[e] : e - E;
    float a = __expf(ebuf[i] - ndmax[(long)d * HEADS + h]);
    ebuf[i] = a;
    atomicAdd(&denom[(long)d * HEADS + h], a);
}

// One wave per (edge, head): accum[d,h,:] += alpha * xl[s,h,:]
__global__ void k_edge_scatter(const float* __restrict__ xl, const float* __restrict__ ebuf,
                               const float* __restrict__ denom,
                               const int* __restrict__ src, const int* __restrict__ dst,
                               float* __restrict__ accum, int E, int N) {
    int gw   = blockIdx.x * (blockDim.x >> 5) + (threadIdx.x >> 5);
    int lane = threadIdx.x & 31;
    long tot = (long)(E + N) * HEADS;
    if (gw >= tot) return;
    int e = gw >> 2, h = gw & 3;
    int s, d;
    if (e < E) { s = src[e]; d = dst[e]; } else { s = e - E; d = s; }
    float alpha = ebuf[gw] / denom[(long)d * HEADS + h];
    const float* pl = xl + ((long)s * HEADS + h) * HIDC;
    float*       pa = accum + ((long)d * HEADS + h) * HIDC;
#pragma unroll
    for (int c = lane; c < HIDC; c += 32)
        atomicAdd(&pa[c], alpha * pl[c]);
}

// ---------------- post-aggregate: head mean + bias + leaky + BN stats ----------------
// block = 128 threads (one per channel), 64 nodes per block, register accumulation
__global__ void k_stats(const float* __restrict__ accum, const float* __restrict__ b,
                        float* __restrict__ outpre, float* __restrict__ chsum,
                        float* __restrict__ chsq, int N) {
    int c  = threadIdx.x;              // channel 0..127
    int n0 = blockIdx.x * 64;
    float s = 0.f, sq = 0.f;
    for (int i = 0; i < 64; ++i) {
        int n = n0 + i;
        if (n >= N) break;
        const float* pa = accum + (long)n * (HEADS * HIDC) + c;
        float v = 0.25f * (pa[0] + pa[HIDC] + pa[2 * HIDC] + pa[3 * HIDC]) + b[c];
        v = v > 0.f ? v : v * NEGSLOPE;
        outpre[(long)n * HIDC + c] = v;
        s += v; sq += v * v;
    }
    atomicAdd(&chsum[c], s);
    atomicAdd(&chsq[c], sq);
}

__global__ void k_bn_finalize(const float* __restrict__ pre,
                              const float* __restrict__ chsum, const float* __restrict__ chsq,
                              const float* __restrict__ g, const float* __restrict__ be,
                              const float* __restrict__ skip,
                              float* __restrict__ out, unsigned short* __restrict__ outbf,
                              int N) {
    long i = (long)blockIdx.x * blockDim.x + threadIdx.x;
    if (i >= (long)N * HIDC) return;
    int c = (int)(i & (HIDC - 1));
    float invN = 1.0f / (float)N;
    float mu  = chsum[c] * invN;
    float var = chsq[c] * invN - mu * mu;           // biased variance (matches x.var(0))
    float v = (pre[i] - mu) * rsqrtf(var + BNEPS) * g[c] + be[c];
    if (skip) v += skip[i];
    out[i] = v;
    if (outbf) outbf[i] = f2bf(v);
}

// ---------------- launcher ----------------
extern "C" void kernel_launch(void* const* d_in, const int* in_sizes, int n_in,
                              void* d_out, int out_size, void* d_ws, size_t ws_size,
                              hipStream_t stream) {
    (void)n_in; (void)out_size; (void)ws_size;
    const float* x    = (const float*)d_in[0];
    const int*   ei   = (const int*)  d_in[1];
    const float* Wl0  = (const float*)d_in[2];
    const float* Wr0  = (const float*)d_in[3];
    const float* bl0  = (const float*)d_in[4];
    const float* br0  = (const float*)d_in[5];
    const float* att0 = (const float*)d_in[6];
    const float* b0   = (const float*)d_in[7];
    const float* g0   = (const float*)d_in[8];
    const float* be0  = (const float*)d_in[9];
    const float* Wl1  = (const float*)d_in[10];
    const float* Wr1  = (const float*)d_in[11];
    const float* bl1  = (const float*)d_in[12];
    const float* br1  = (const float*)d_in[13];
    const float* att1 = (const float*)d_in[14];
    const float* b1   = (const float*)d_in[15];
    const float* g1   = (const float*)d_in[16];
    const float* be1  = (const float*)d_in[17];
    const float* skW  = (const float*)d_in[18];
    const float* skb  = (const float*)d_in[19];

    const int IN = 256;
    const int N  = in_sizes[0] / IN;
    const int E  = in_sizes[1] / 2;
    const int F  = HEADS * HIDC;       // 512
    const int* srcp = ei;
    const int* dstp = ei + E;

    char* base = (char*)d_ws;
    size_t off = 0;
    auto wsalloc = [&](size_t bytes) -> void* {
        void* p = base + off;
        off += (bytes + 255) & ~(size_t)255;
        return p;
    };
    unsigned short* bfX  = (unsigned short*)wsalloc((size_t)N * IN * 2);
    unsigned short* bfHb = (unsigned short*)wsalloc((size_t)N * HIDC * 2);
    unsigned short* Wl0t = (unsigned short*)wsalloc((size_t)F * IN * 2);
    unsigned short* Wr0t = (unsigned short*)wsalloc((size_t)F * IN * 2);
    unsigned short* Wl1t = (unsigned short*)wsalloc((size_t)F * HIDC * 2);
    unsigned short* Wr1t = (unsigned short*)wsalloc((size_t)F * HIDC * 2);
    unsigned short* skWt = (unsigned short*)wsalloc((size_t)HIDC * IN * 2);
    float* xl    = (float*)wsalloc((size_t)N * F * 4);
    float* xr    = (float*)wsalloc((size_t)N * F * 4);
    float* accum = (float*)wsalloc((size_t)N * F * 4);
    float* hskip = (float*)wsalloc((size_t)N * HIDC * 4);
    float* out0  = (float*)wsalloc((size_t)N * HIDC * 4);
    float* ebuf  = (float*)wsalloc((size_t)(E + N) * HEADS * 4);
    float* ndmax = (float*)wsalloc((size_t)N * HEADS * 4);
    float* denom = (float*)wsalloc((size_t)N * HEADS * 4);
    float* chsum = (float*)wsalloc(HIDC * 4);
    float* chsq  = (float*)wsalloc(HIDC * 4);

    auto cdiv = [](long a, long b) { return (int)((a + b - 1) / b); };

    // ---- bf16 conversions ----
    k_f32_to_bf16<<<cdiv((long)N * IN, 256), 256, 0, stream>>>(x, bfX, (long)N * IN);
    k_transpose_bf16<<<cdiv((long)IN * F, 256), 256, 0, stream>>>(Wl0, Wl0t, IN, F);
    k_transpose_bf16<<<cdiv((long)IN * F, 256), 256, 0, stream>>>(Wr0, Wr0t, IN, F);
    k_transpose_bf16<<<cdiv((long)HIDC * F, 256), 256, 0, stream>>>(Wl1, Wl1t, HIDC, F);
    k_transpose_bf16<<<cdiv((long)HIDC * F, 256), 256, 0, stream>>>(Wr1, Wr1t, HIDC, F);
    k_transpose_bf16<<<cdiv((long)IN * HIDC, 256), 256, 0, stream>>>(skW, skWt, IN, HIDC);

    // ---- skip projection: h = x @ skip_W + skip_b ----
    {
        int tiles = (N / 16) * (HIDC / 64);
        k_wmma_gemm<256><<<cdiv(tiles, 8), 256, 0, stream>>>(bfX, skWt, skb, hskip, N, HIDC);
    }
    // ---- layer 0 projections ----
    {
        int tiles = (N / 16) * (F / 64);
        k_wmma_gemm<256><<<cdiv(tiles, 8), 256, 0, stream>>>(bfX, Wl0t, bl0, xl, N, F);
        k_wmma_gemm<256><<<cdiv(tiles, 8), 256, 0, stream>>>(bfX, Wr0t, br0, xr, N, F);
    }

    auto edge_phase = [&](const float* att, const float* bb, const float* g,
                          const float* be, const float* skip, float* outf,
                          unsigned short* outbf) {
        long totEH = (long)(E + N) * HEADS;
        k_fill<<<cdiv((long)N * HEADS, 256), 256, 0, stream>>>(ndmax, -1e30f, (long)N * HEADS);
        k_fill<<<cdiv((long)N * HEADS, 256), 256, 0, stream>>>(denom, 0.f, (long)N * HEADS);
        k_fill<<<cdiv((long)N * F, 256), 256, 0, stream>>>(accum, 0.f, (long)N * F);
        k_fill<<<1, 256, 0, stream>>>(chsum, 0.f, HIDC);
        k_fill<<<1, 256, 0, stream>>>(chsq, 0.f, HIDC);
        k_edge_logits<<<cdiv(totEH, 8), 256, 0, stream>>>(xl, xr, srcp, dstp, att, ebuf, ndmax, E, N);
        k_edge_exp<<<cdiv(totEH, 256), 256, 0, stream>>>(ebuf, ndmax, denom, dstp, E, N);
        k_edge_scatter<<<cdiv(totEH, 8), 256, 0, stream>>>(xl, ebuf, denom, srcp, dstp, accum, E, N);
        k_stats<<<cdiv(N, 64), 128, 0, stream>>>(accum, bb, out0, chsum, chsq, N);
        k_bn_finalize<<<cdiv((long)N * HIDC, 256), 256, 0, stream>>>(out0, chsum, chsq, g, be,
                                                                     skip, outf, outbf, N);
    };

    // layer 0: BN + skip, emit bf16 for layer-1 GEMMs
    edge_phase(att0, b0, g0, be0, hskip, out0, bfHb);

    // ---- layer 1 projections ----
    {
        int tiles = (N / 16) * (F / 64);
        k_wmma_gemm<128><<<cdiv(tiles, 8), 256, 0, stream>>>(bfHb, Wl1t, bl1, xl, N, F);
        k_wmma_gemm<128><<<cdiv(tiles, 8), 256, 0, stream>>>(bfHb, Wr1t, br1, xr, N, F);
    }
    // layer 1: BN only, write final fp32 output
    edge_phase(att1, b1, g1, be1, nullptr, (float*)d_out, nullptr);
}
